// ConvNet_STDP_90254442758741
// MI455X (gfx1250) — matrix operand order repaired and checked by reading, not compiled, not run on hardware
//
#include <hip/hip_runtime.h>
#include <hip/hip_bf16.h>

typedef __attribute__((ext_vector_type(16))) _Float16 v16h;
typedef __attribute__((ext_vector_type(8)))  _Float16 v8h;
typedef __attribute__((ext_vector_type(8)))  float    v8f;

#define DT_TAU_MEM 0.1f   // DT * TAU_MEM_INV
#define DT_TAU_SYN 0.2f   // DT * TAU_SYN_INV
#define DT_TAU_TR  0.02f  // DT * TAU_PRE/POST_INV
#define ETA_P 0.004f
#define ETA_M 0.003f

// ---------------- DoG filter + pre-synaptic trace (layer1) ----------------
// z[b,2,60,60]: ch0 = conv(x, g1-g2) + (b1-b2), ch1 = -ch0 ; tp1 = 0.98*tp1 + 0.02*z
__global__ void k_dog(const float* __restrict__ x, const float* __restrict__ b1,
                      const float* __restrict__ b2, float* __restrict__ z,
                      float* __restrict__ tp1, int B)
{
    int id = blockIdx.x * blockDim.x + threadIdx.x;
    if (id >= B * 3600) return;
    int b = id / 3600, p = id % 3600, oy = p / 60, ox = p % 60;

    float s1 = 0.f, s2 = 0.f;
    #pragma unroll
    for (int d = 0; d < 25; ++d) {
        float dy = (float)(d / 5) - 2.f, dx = (float)(d % 5) - 2.f;
        float r2 = dy * dy + dx * dx;
        s1 += __expf(-r2 * 0.5f);
        s2 += __expf(-r2 * 0.125f);
    }
    float w[25];
    #pragma unroll
    for (int d = 0; d < 25; ++d) {
        float dy = (float)(d / 5) - 2.f, dx = (float)(d % 5) - 2.f;
        float r2 = dy * dy + dx * dx;
        w[d] = __expf(-r2 * 0.5f) / s1 - __expf(-r2 * 0.125f) / s2;
    }
    const float* xb = x + (size_t)b * 4096;
    float acc = 0.f;
    #pragma unroll
    for (int u = 0; u < 5; ++u)
        #pragma unroll
        for (int v = 0; v < 5; ++v)
            acc += xb[(oy + u) * 64 + ox + v] * w[u * 5 + v];
    float d0 = acc + (b1[0] - b2[0]);
    size_t i0 = (size_t)b * 7200 + p, i1 = i0 + 3600;
    z[i0] = d0;  z[i1] = -d0;
    tp1[i0] = tp1[i0] * (1.f - DT_TAU_TR) + DT_TAU_TR * d0;
    tp1[i1] = tp1[i1] * (1.f - DT_TAU_TR) + DT_TAU_TR * (-d0);
}

// ---------------- im2col layer1, row-major [R1][64] (forward GEMM A) ----------------
__global__ void k_im2col1_rm(const float* __restrict__ z, _Float16* __restrict__ Uz, int B)
{
    int id = blockIdx.x * blockDim.x + threadIdx.x;
    int R1 = B * 3136;
    if (id >= R1 * 64) return;
    int r = id >> 6, i = id & 63;
    int b = r / 3136, p = r % 3136, oy = p / 56, ox = p % 56;
    float vz = 0.f;
    if (i < 50) {
        int c = i / 25, q = i % 25, kh = q / 5, kw = q % 5;
        vz = z[(size_t)b * 7200 + (size_t)c * 3600 + (size_t)(oy + kh) * 60 + (ox + kw)];
    }
    Uz[id] = (_Float16)vz;
}

// ---------------- im2col layer1, col-major [64][R1] (grad GEMM B operands) ----------------
__global__ void k_im2col1_cm(const float* __restrict__ z, const float* __restrict__ tp1,
                             _Float16* __restrict__ UzT, _Float16* __restrict__ UtT, int B)
{
    int id = blockIdx.x * blockDim.x + threadIdx.x;
    int R1 = B * 3136;
    if (id >= 64 * R1) return;
    int i = id / R1, r = id % R1;
    int b = r / 3136, p = r % 3136, oy = p / 56, ox = p % 56;
    float vz = 0.f, vt = 0.f;
    if (i < 50) {
        int c = i / 25, q = i % 25, kh = q / 5, kw = q % 5;
        size_t s = (size_t)b * 7200 + (size_t)c * 3600 + (size_t)(oy + kh) * 60 + (ox + kw);
        vz = z[s]; vt = tp1[s];
    }
    UzT[id] = (_Float16)vz;
    UtT[id] = (_Float16)vt;
}

// ---------------- Forward GEMM: Ct[Np][M] = scale * A[M,Kp] * W[Np,Kp]^T ----------------
// One wave per 16x16 tile; transposed D store -> 8 consecutive floats per lane.
__global__ void __launch_bounds__(256) k_gemm_fwd(const _Float16* __restrict__ A,
                                                  const _Float16* __restrict__ W,
                                                  float* __restrict__ Ct,
                                                  int M, int Np, int Kp, float scale)
{
    int lane = threadIdx.x & 31;
    int half = lane >> 4;
    int mrow = lane & 15;
    int wid  = (blockIdx.x << 3) + (threadIdx.x >> 5);
    int tilesM = M >> 4, tilesN = Np >> 4;
    if (wid >= tilesM * tilesN) return;   // wave-uniform: EXEC stays all-ones
    int tm = (wid % tilesM) << 4;
    int tn = (wid / tilesM) << 4;

    v8f acc = {};
    const _Float16* arow = A + (size_t)(tm + mrow) * Kp + half * 16;
    const _Float16* brow = W + (size_t)(tn + mrow) * Kp + half * 16;
    for (int kk = 0; kk < Kp; kk += 32) {
        v8h a0 = *(const v8h*)(arow + kk);
        v8h a1 = *(const v8h*)(arow + kk + 8);
        v16h a, bm;
        #pragma unroll
        for (int e = 0; e < 8; ++e) { a[e] = a0[e]; a[8 + e] = a1[e]; }
        bm = *(const v16h*)(brow + kk);
        acc = __builtin_amdgcn_wmma_f32_16x16x32_f16(false, a, false, bm,
                                                     (short)0, acc, false, false);
    }
    v8f outv;
    #pragma unroll
    for (int r = 0; r < 8; ++r) outv[r] = scale * acc[r];
    *(v8f*)(Ct + (size_t)(tn + mrow) * M + tm + half * 8) = outv;
}

// ---------------- LIF step + post-synaptic trace (channel-major layouts) ----------------
__global__ void k_lif_trace(const float* __restrict__ pre, float* __restrict__ v,
                            float* __restrict__ cur, _Float16* __restrict__ zT,
                            float* __restrict__ tq, _Float16* __restrict__ tqT,
                            int n, float th)
{
    int id = blockIdx.x * blockDim.x + threadIdx.x;
    if (id >= n) return;
    float vv = v[id], ii = cur[id];
    float vdec = vv + DT_TAU_MEM * (ii - vv);
    float idec = (1.f - DT_TAU_SYN) * ii;
    float zz = (vdec - th > 0.f) ? 1.f : 0.f;
    v[id]   = (1.f - zz) * vdec;
    cur[id] = idec + pre[id];
    zT[id]  = (_Float16)zz;
    float t = tq[id] * (1.f - DT_TAU_TR) + DT_TAU_TR * zz;
    tq[id]  = t;
    tqT[id] = (_Float16)t;
}

// ---------------- 2x2 maxpool + layer2 pre-synaptic trace ([C][R] layouts) ----------------
__global__ void k_pool_trace(const _Float16* __restrict__ z2T, float* __restrict__ z3T,
                             float* __restrict__ tp2, int B)
{
    int id = blockIdx.x * blockDim.x + threadIdx.x;
    int R3 = B * 784, R1 = B * 3136;
    if (id >= 32 * R3) return;
    int c = id / R3, p = id % R3;
    int b = p / 784, q = p % 784, oy = q / 28, ox = q % 28;
    size_t base = (size_t)c * R1 + (size_t)b * 3136 + (size_t)(oy * 2) * 56 + ox * 2;
    float v00 = (float)z2T[base];
    float v01 = (float)z2T[base + 1];
    float v10 = (float)z2T[base + 56];
    float v11 = (float)z2T[base + 57];
    float m = fmaxf(fmaxf(v00, v01), fmaxf(v10, v11));
    z3T[id] = m;
    tp2[id] = tp2[id] * (1.f - DT_TAU_TR) + DT_TAU_TR * m;
}

// ---------------- im2col layer2, row-major [R2][768] (forward GEMM A) ----------------
__global__ void k_im2col2_rm(const float* __restrict__ z3T, _Float16* __restrict__ Az, int B)
{
    int id = blockIdx.x * blockDim.x + threadIdx.x;
    int R2 = B * 576, R3 = B * 784;
    if (id >= R2 * 768) return;
    int r = id / 768, i = id % 768;
    int b = r / 576, p = r % 576, oy = p / 24, ox = p % 24;
    float vz = 0.f;
    if (i < 750) {
        int c = i / 25, q = i % 25, kh = q / 5, kw = q % 5;
        vz = z3T[(size_t)c * R3 + (size_t)b * 784 + (size_t)(oy + kh) * 28 + (ox + kw)];
    }
    Az[id] = (_Float16)vz;
}

// ---------------- im2col layer2, col-major [768][R2] (grad GEMM B operands) ----------------
__global__ void k_im2col2_cm(const float* __restrict__ z3T, const float* __restrict__ tp2,
                             _Float16* __restrict__ AzT, _Float16* __restrict__ AtT, int B)
{
    int id = blockIdx.x * blockDim.x + threadIdx.x;
    int R2 = B * 576, R3 = B * 784;
    if (id >= 768 * R2) return;
    int i = id / R2, r = id % R2;
    int b = r / 576, p = r % 576, oy = p / 24, ox = p % 24;
    float vz = 0.f, vt = 0.f;
    if (i < 750) {
        int c = i / 25, q = i % 25, kh = q / 5, kw = q % 5;
        size_t s = (size_t)c * R3 + (size_t)b * 784 + (size_t)(oy + kh) * 28 + (ox + kw);
        vz = z3T[s]; vt = tp2[s];
    }
    AzT[id] = (_Float16)vz;
    AtT[id] = (_Float16)vt;
}

// ---------------- STDP weight-grad GEMM (split-K, vector loads, f32 atomics) ----------------
// dw[j,i] += sum_r Zt[j][r] * Ut[i][r]   (Zt: [Nz][R] f16, Ut: [Kp][R] f16)
__global__ void __launch_bounds__(256) k_gemm_grad(const _Float16* __restrict__ Zt,
                                                   const _Float16* __restrict__ Ut,
                                                   float* __restrict__ dw,
                                                   int R, int Nz, int Kp,
                                                   int chunkR, int totalWaves)
{
    int lane = threadIdx.x & 31;
    int half = lane >> 4;
    int mrow = lane & 15;
    int wid = (blockIdx.x << 3) + (threadIdx.x >> 5);
    if (wid >= totalWaves) return;   // wave-uniform
    int tilesJ = Nz >> 4, tilesI = Kp >> 4;
    int tpc = tilesJ * tilesI;
    int cid = wid / tpc, tix = wid % tpc;
    int tj = (tix % tilesJ) << 4;
    int ti = (tix / tilesJ) << 4;
    int r0 = cid * chunkR;
    int r1 = r0 + chunkR; if (r1 > R) r1 = R;

    v8f acc = {};
    const _Float16* arow = Zt + (size_t)(tj + mrow) * R + half * 16;
    const _Float16* brow = Ut + (size_t)(ti + mrow) * R + half * 16;
    for (int r = r0; r < r1; r += 32) {
        v8h a0 = *(const v8h*)(arow + r);
        v8h a1 = *(const v8h*)(arow + r + 8);
        v16h a, bm;
        #pragma unroll
        for (int e = 0; e < 8; ++e) { a[e] = a0[e]; a[8 + e] = a1[e]; }
        bm = *(const v16h*)(brow + r);
        acc = __builtin_amdgcn_wmma_f32_16x16x32_f16(false, a, false, bm,
                                                     (short)0, acc, false, false);
    }
    #pragma unroll
    for (int rr = 0; rr < 8; ++rr)
        atomicAdd(&dw[(size_t)(tj + half * 8 + rr) * Kp + ti + mrow], acc[rr]);
}

// ---------------- pack input weights into padded f32 master + f16 copy ----------------
__global__ void k_packw(const float* __restrict__ src, float* __restrict__ dstf,
                        _Float16* __restrict__ dsth, int J, int I, int Kp, int total)
{
    int id = blockIdx.x * blockDim.x + threadIdx.x;
    if (id >= total) return;
    int j = id / Kp, i = id % Kp;
    float v = (j < J && i < I) ? src[(size_t)j * I + i] : 0.f;
    dstf[id] = v;
    dsth[id] = (_Float16)v;
}

// ---------------- STDP multiplicative-relu update, hard bounds [0,1] ----------------
__global__ void k_update_w(float* __restrict__ w, _Float16* __restrict__ wh,
                           const float* __restrict__ dwP, const float* __restrict__ dwM,
                           int J, int I, int Kp, int total)
{
    int id = blockIdx.x * blockDim.x + threadIdx.x;
    if (id >= total) return;
    int j = id / Kp, i = id % Kp;
    if (j < J && i < I) {
        float v = w[id];
        float up = ETA_P * fmaxf(1.f - v, 0.f) * dwP[id];
        float dn = ETA_M * fmaxf(v, 0.f) * dwM[id];
        v = v + up - dn;
        v = fminf(fmaxf(v, 0.f), 1.f);
        w[id] = v;
        wh[id] = (_Float16)v;
    } else {
        w[id] = 0.f;
        wh[id] = (_Float16)0.f;
    }
}

// ---------------- spatial max over 24x24 -> g[3200] (contiguous per channel) ----------------
__global__ void k_gmax(const _Float16* __restrict__ z4T, float* __restrict__ g, int B)
{
    int k = blockIdx.x;              // 0..3199
    int b = k / 100, c = k % 100;
    int R2 = B * 576;
    __shared__ float red[64];
    const _Float16* src = z4T + (size_t)c * R2 + (size_t)b * 576;
    float m = 0.f;
    for (int l = threadIdx.x; l < 576; l += 64)
        m = fmaxf(m, (float)src[l]);
    red[threadIdx.x] = m;
    __syncthreads();
    for (int s = 32; s > 0; s >>= 1) {
        if (threadIdx.x < s) red[threadIdx.x] = fmaxf(red[threadIdx.x], red[threadIdx.x + s]);
        __syncthreads();
    }
    if (threadIdx.x == 0) g[k] = red[0];
}

// ---------------- head: fc1 GEMV + LIF(th=1) + LILinearCell + voltage write ----------------
__global__ void k_head(const float* __restrict__ g, const float* __restrict__ fc1w,
                       const float* __restrict__ fc1b, const float* __restrict__ outw,
                       float* __restrict__ v2, float* __restrict__ i2,
                       float* __restrict__ vo, float* __restrict__ io,
                       float* __restrict__ out, int t, int B)
{
    __shared__ float h[500];
    __shared__ float vol[10];
    int tid = threadIdx.x;
    if (tid < 500) {
        float acc = fc1b[tid];
        const float* wr = fc1w + (size_t)tid * 3200;
        for (int k = 0; k < 3200; ++k) acc += g[k] * wr[k];
        float v = v2[tid], cu = i2[tid];
        float vdec = v + DT_TAU_MEM * (cu - v);
        float idec = (1.f - DT_TAU_SYN) * cu;
        float z = (vdec - 1.f > 0.f) ? 1.f : 0.f;
        v2[tid] = (1.f - z) * vdec;
        i2[tid] = idec + acc;
        h[tid] = z;
    }
    __syncthreads();
    if (tid < 10) {
        float acc = 0.f;
        const float* wr = outw + (size_t)tid * 500;
        for (int c = 0; c < 500; ++c) acc += fmaxf(h[c], 0.f) * wr[c];
        float ij = io[tid] + acc;
        float v = vo[tid] + DT_TAU_MEM * (ij - vo[tid]);
        vo[tid] = v;
        io[tid] = (1.f - DT_TAU_SYN) * ij;
        vol[tid] = v;
    }
    __syncthreads();
    for (int idx = tid; idx < B * 10; idx += blockDim.x)
        out[(size_t)t * B * 10 + idx] = vol[idx % 10];
}

extern "C" void kernel_launch(void* const* d_in, const int* in_sizes, int n_in,
                              void* d_out, int out_size, void* d_ws, size_t ws_size,
                              hipStream_t stream)
{
    (void)in_sizes; (void)n_in; (void)out_size; (void)ws_size;
    const float* x    = (const float*)d_in[0];
    const float* b1   = (const float*)d_in[1];
    const float* b2   = (const float*)d_in[2];
    const float* w1in = (const float*)d_in[3];
    const float* w2in = (const float*)d_in[4];
    const float* fc1w = (const float*)d_in[5];
    const float* fc1b = (const float*)d_in[6];
    const float* outw = (const float*)d_in[7];
    float* out = (float*)d_out;

    const int T = 32, B = 32;
    const int R1 = B * 3136;   // conv1 output positions: 100352
    const int R3 = B * 784;    // pooled positions:       25088
    const int R2 = B * 576;    // conv2 output positions: 18432

    size_t off = 0;
    auto alloc = [&](size_t bytes) -> char* {
        char* p = (char*)d_ws + off;
        off += (bytes + 255) & ~(size_t)255;
        return p;
    };
    // weights (padded masters + f16)
    float*    w1s  = (float*)   alloc((size_t)32 * 64 * 4);
    _Float16* W1h  = (_Float16*)alloc((size_t)32 * 64 * 2);
    float*    w2s  = (float*)   alloc((size_t)112 * 768 * 4);
    _Float16* W2h  = (_Float16*)alloc((size_t)112 * 768 * 2);
    // dw accumulators (contiguous so one memset clears all four)
    float* dwP1 = (float*)alloc((size_t)32 * 64 * 4);
    float* dwM1 = (float*)alloc((size_t)32 * 64 * 4);
    float* dwP2 = (float*)alloc((size_t)112 * 768 * 4);
    float* dwM2 = (float*)alloc((size_t)112 * 768 * 4);
    size_t dwBytes = ((size_t)32 * 64 * 4) * 2 + ((size_t)112 * 768 * 4) * 2;
    // LIF / trace state (channel-major [C][M] layouts)
    float* v0  = (float*)alloc((size_t)32 * R1 * 4);
    float* i0  = (float*)alloc((size_t)32 * R1 * 4);
    float* tq1 = (float*)alloc((size_t)32 * R1 * 4);
    float* tp1 = (float*)alloc((size_t)B * 7200 * 4);
    float* v1  = (float*)alloc((size_t)112 * R2 * 4);
    float* i1  = (float*)alloc((size_t)112 * R2 * 4);
    float* tq2 = (float*)alloc((size_t)112 * R2 * 4);
    float* tp2 = (float*)alloc((size_t)32 * R3 * 4);
    float* v2  = (float*)alloc(512 * 4);
    float* i2  = (float*)alloc(512 * 4);
    float* vo  = (float*)alloc(64 * 4);
    float* io  = (float*)alloc(64 * 4);
    // per-step scratch
    float*    z      = (float*)   alloc((size_t)B * 7200 * 4);
    float*    pre1   = (float*)   alloc((size_t)32 * R1 * 4);   // [32][R1]
    _Float16* Uz     = (_Float16*)alloc((size_t)R1 * 64 * 2);   // [R1][64]
    _Float16* UzT    = (_Float16*)alloc((size_t)64 * R1 * 2);   // [64][R1]
    _Float16* UtT    = (_Float16*)alloc((size_t)64 * R1 * 2);   // [64][R1]
    _Float16* z2T    = (_Float16*)alloc((size_t)32 * R1 * 2);   // [32][R1]
    _Float16* tq1T   = (_Float16*)alloc((size_t)32 * R1 * 2);
    float*    z3T    = (float*)   alloc((size_t)32 * R3 * 4);   // [32][R3]
    _Float16* A2z    = (_Float16*)alloc((size_t)R2 * 768 * 2);  // [R2][768]
    _Float16* A2zT   = (_Float16*)alloc((size_t)768 * R2 * 2);  // [768][R2]
    _Float16* A2tT   = (_Float16*)alloc((size_t)768 * R2 * 2);
    float*    pre2   = (float*)   alloc((size_t)112 * R2 * 4);  // [112][R2]
    _Float16* z4T    = (_Float16*)alloc((size_t)112 * R2 * 2);  // [112][R2]
    _Float16* tq2T   = (_Float16*)alloc((size_t)112 * R2 * 2);
    float*    gbuf   = (float*)   alloc(3200 * 4);

    // deterministic per-call init of all state + scratch
    hipMemsetAsync(d_ws, 0, off, stream);
    k_packw<<<(32 * 64 + 255) / 256, 256, 0, stream>>>(w1in, w1s, W1h, 30, 50, 64, 32 * 64);
    k_packw<<<(112 * 768 + 255) / 256, 256, 0, stream>>>(w2in, w2s, W2h, 100, 750, 768, 112 * 768);

    for (int t = 0; t < T; ++t) {
        const float* xt = x + (size_t)t * B * 4096;
        k_dog<<<(B * 3600 + 255) / 256, 256, 0, stream>>>(xt, b1, b2, z, tp1, B);
        k_im2col1_rm<<<(R1 * 64 + 255) / 256, 256, 0, stream>>>(z, Uz, B);
        k_im2col1_cm<<<(64 * R1 + 255) / 256, 256, 0, stream>>>(z, tp1, UzT, UtT, B);
        // conv1: [R1,64] x [32,64]^T -> pre1 [32][R1]
        {
            int tiles = (R1 / 16) * (32 / 16);
            k_gemm_fwd<<<(tiles + 7) / 8, 256, 0, stream>>>(Uz, W1h, pre1, R1, 32, 64, 1.0f);
        }
        k_lif_trace<<<(32 * R1 + 255) / 256, 256, 0, stream>>>(pre1, v0, i0, z2T, tq1, tq1T,
                                                               32 * R1, 15.0f);
        hipMemsetAsync(dwP1, 0, dwBytes, stream);  // clears dwP1,dwM1,dwP2,dwM2
        // stdp1 grads: K = R1
        {
            int chunk = 2048;
            int chunks = (R1 + chunk - 1) / chunk;
            int tw = 2 * 4 * chunks;  // tilesJ=2, tilesI=4
            k_gemm_grad<<<(tw + 7) / 8, 256, 0, stream>>>(z2T, UtT, dwP1, R1, 32, 64, chunk, tw);
            k_gemm_grad<<<(tw + 7) / 8, 256, 0, stream>>>(tq1T, UzT, dwM1, R1, 32, 64, chunk, tw);
        }
        k_update_w<<<(32 * 64 + 255) / 256, 256, 0, stream>>>(w1s, W1h, dwP1, dwM1, 30, 50, 64, 32 * 64);
        k_pool_trace<<<(32 * R3 + 255) / 256, 256, 0, stream>>>(z2T, z3T, tp2, B);
        k_im2col2_rm<<<(R2 * 768 + 255) / 256, 256, 0, stream>>>(z3T, A2z, B);
        k_im2col2_cm<<<(768 * R2 + 255) / 256, 256, 0, stream>>>(z3T, tp2, A2zT, A2tT, B);
        // conv2: [R2,768] x [112,768]^T, scaled by 10 -> pre2 [112][R2]
        {
            int tiles = (R2 / 16) * (112 / 16);
            k_gemm_fwd<<<(tiles + 7) / 8, 256, 0, stream>>>(A2z, W2h, pre2, R2, 112, 768, 10.0f);
        }
        k_lif_trace<<<(112 * R2 + 255) / 256, 256, 0, stream>>>(pre2, v1, i1, z4T, tq2, tq2T,
                                                                112 * R2, 10.0f);
        // stdp2 grads: K = R2
        {
            int chunk = 2048;
            int chunks = (R2 + chunk - 1) / chunk;
            int tw = 7 * 48 * chunks;  // tilesJ=7, tilesI=48
            k_gemm_grad<<<(tw + 7) / 8, 256, 0, stream>>>(z4T, A2tT, dwP2, R2, 112, 768, chunk, tw);
            k_gemm_grad<<<(tw + 7) / 8, 256, 0, stream>>>(tq2T, A2zT, dwM2, R2, 112, 768, chunk, tw);
        }
        k_update_w<<<(112 * 768 + 255) / 256, 256, 0, stream>>>(w2s, W2h, dwP2, dwM2, 100, 750, 768,
                                                                112 * 768);
        k_gmax<<<3200, 64, 0, stream>>>(z4T, gbuf, B);
        k_head<<<1, 512, 0, stream>>>(gbuf, fc1w, fc1b, outw, v2, i2, vo, io, out, t, B);
    }
}
